// EvaLinearAttention_40656160424185
// MI455X (gfx1250) — matrix operand order repaired and checked by reading, not compiled, not run on hardware
//
#include <hip/hip_runtime.h>
#include <hip/hip_bf16.h>

typedef __attribute__((ext_vector_type(16))) _Float16 v16h;
typedef __attribute__((ext_vector_type(8)))  _Float16 v8h;
typedef __attribute__((ext_vector_type(8)))  float    v8f;

// ---------------------------------------------------------------------------
// WMMA helpers (CDNA5 / gfx1250, wave32)
// D = A(16x32 f16) x B(32x16 f16) + C(16x16 f32)
// ---------------------------------------------------------------------------
static __device__ __forceinline__ v8f wmma16(v16h a, v16h b, v8f c) {
  return __builtin_amdgcn_wmma_f32_16x16x32_f16(
      /*neg_a=*/false, a, /*neg_b=*/false, b,
      /*c_mod=*/(short)0, c, /*reuse_a=*/false, /*reuse_b=*/false);
}

// A-matrix fragment (16x32, row-major source, lda in halves).
// ISA layout: lanes 0-15 hold M=lane, K=0..7 (v0..3) and K=16..23 (v4..7);
// lanes 16-31 hold K=8..15 and K=24..31.  Two contiguous 16B loads per lane.
static __device__ __forceinline__ v16h load_a_frag(const _Float16* A, int lda,
                                                   int row_base, int k0,
                                                   int lane, int Mmax) {
  int m = row_base + (lane & 15);
  if (m >= Mmax) m = Mmax - 1;   // clamp: OOB rows computed but never stored
  int kb = k0 + ((lane >> 4) << 3);
  const v8h* p0 = (const v8h*)(A + (size_t)m * lda + kb);
  const v8h* p1 = (const v8h*)(A + (size_t)m * lda + kb + 16);
  v8h lo = *p0, hi = *p1;
  v16h r;
#pragma unroll
  for (int i = 0; i < 8; ++i) { r[i] = lo[i]; r[i + 8] = hi[i]; }
  return r;
}

// B-matrix fragment from transposed tile Bt[n][k] (ldb in halves).
// Lanes 0-15: N=lane, K=0..7,16..23; lanes 16-31: N=lane-16, K=8..15,24..31.
// Two 16B reads per lane, at K and K+16 (2x ds_load_b128).
static __device__ __forceinline__ v16h load_bt_frag(const _Float16* Bt, int ldb,
                                                    int n_base, int k0, int lane) {
  int n = n_base + (lane & 15);
  int kb = k0 + ((lane >> 4) << 3);
  const v8h* p = (const v8h*)(Bt + (size_t)n * ldb + kb);
  const v8h* q = (const v8h*)(Bt + (size_t)n * ldb + kb + 16);
  v8h lo = p[0], hi = q[0];
  v16h r;
#pragma unroll
  for (int i = 0; i < 8; ++i) { r[i] = lo[i]; r[i + 8] = hi[i]; }
  return r;
}

// Generic LDS pointer -> 32-bit LDS byte address (generic LDS addr = aperture
// high bits | lds offset in low 32 bits; see ISA 10.2).
static __device__ __forceinline__ unsigned lds_addr32(const void* p) {
  return (unsigned)(unsigned long long)(uintptr_t)p;
}

// ---------------------------------------------------------------------------
// Elementwise utility kernels
// ---------------------------------------------------------------------------
__global__ void cvt_f32_f16_kernel(const float* __restrict__ src,
                                   _Float16* __restrict__ dst, size_t n) {
  size_t i = (size_t)blockIdx.x * blockDim.x + threadIdx.x;
  size_t stride = (size_t)gridDim.x * blockDim.x;
  for (; i < n; i += stride) dst[i] = (_Float16)src[i];
}

__global__ void zero_f32_kernel(float* __restrict__ dst, size_t n) {
  size_t i = (size_t)blockIdx.x * blockDim.x + threadIdx.x;
  size_t stride = (size_t)gridDim.x * blockDim.x;
  for (; i < n; i += stride) dst[i] = 0.f;
}

// bias = concat(q_bias, zeros(C), v_bias), length 3C
__global__ void build_bias_kernel(const float* __restrict__ q_bias,
                                  const float* __restrict__ v_bias,
                                  float* __restrict__ bias, int C) {
  int i = blockIdx.x * blockDim.x + threadIdx.x;
  if (i >= 3 * C) return;
  float v = 0.f;
  if (i < C) v = q_bias[i];
  else if (i >= 2 * C) v = v_bias[i - 2 * C];
  bias[i] = v;
}

// ---------------------------------------------------------------------------
// Tiled WMMA GEMM:  Cout[M,Nc] = A[M,K] @ W[Nc,K]^T + bias[Nc]
// Block: 256 thr = 8 waves (4 M-waves x 2 N-waves). Block tile 128x128,
// wave tile 32x64 (2x4 WMMA tiles).
// B slab staged transposed in LDS via double-buffered ASYNC global->LDS DMA
// (global_load_async_to_lds_b128 + s_wait_asynccnt): slab ks+1 streams into
// buffer ^1 while WMMAs consume buffer cur. Async ops complete in order, so
// s_wait_asynccnt 2 (the 2 newly issued ops may stay outstanding) guarantees
// the current slab is resident.
// ---------------------------------------------------------------------------
template <bool F16OUT>
__global__ __launch_bounds__(256) void gemm_xWT_kernel(
    const _Float16* __restrict__ A, const _Float16* __restrict__ W,
    const float* __restrict__ bias, void* __restrict__ Cout,
    int M, int Nc, int K) {
  const int m0   = blockIdx.x * 128;
  const int n0   = blockIdx.y * 128;
  const int tid  = threadIdx.x;
  const int wave = tid >> 5;
  const int lane = tid & 31;
  const int wm0  = m0 + (wave & 3) * 32;  // wave M origin (global)
  const int wn0  = (wave >> 2) * 64;      // wave N origin (within tile)

  __shared__ __align__(128) _Float16 Bt[2][128 * 32];  // double-buffered k-slab

  v8f acc[2][4];
  const v8f zac = {0.f, 0.f, 0.f, 0.f, 0.f, 0.f, 0.f, 0.f};
#pragma unroll
  for (int i = 0; i < 2; ++i)
#pragma unroll
    for (int j = 0; j < 4; ++j) acc[i][j] = zac;

  // Async-stage one 128x32 half slab: 512 chunks of 16B; thread -> chunks
  // {tid, tid+256}. Chunk c: W row n0 + (c>>2), halves (c&3)*8.. ; LDS byte
  // offset c*16 (row-major ld=32 halves = 64B/row).
  auto issue_slab = [&](int ks, int buf) {
#pragma unroll
    for (int cc = 0; cc < 2; ++cc) {
      int c = tid + (cc << 8);
      const _Float16* g =
          W + (size_t)(n0 + (c >> 2)) * K + (size_t)ks * 32 + ((c & 3) << 3);
      unsigned laddr = lds_addr32(&Bt[buf][0]) + (unsigned)c * 16u;
      unsigned long long gaddr = (unsigned long long)(uintptr_t)g;
      asm volatile("global_load_async_to_lds_b128 %0, %1, off"
                   :: "v"(laddr), "v"(gaddr)
                   : "memory");
    }
  };

  const int KSTEPS = K >> 5;
  issue_slab(0, 0);

  for (int ks = 0; ks < KSTEPS; ++ks) {
    const int cur = ks & 1;
    const int k0 = ks << 5;
    if (ks + 1 < KSTEPS) {
      issue_slab(ks + 1, cur ^ 1);                       // overlap next slab DMA
      asm volatile("s_wait_asynccnt 0x2" ::: "memory");  // slab ks resident
    } else {
      asm volatile("s_wait_asynccnt 0x0" ::: "memory");
    }
    __syncthreads();  // all waves' async ops for slab ks complete

    if (ks + 1 < KSTEPS)  // warm L2 for next A slab
      __builtin_prefetch(A + (size_t)(wm0 + (lane & 15)) * K + k0 + 32, 0, 1);

    v16h bfrag[4];
#pragma unroll
    for (int j = 0; j < 4; ++j)
      bfrag[j] = load_bt_frag(&Bt[cur][0], 32, wn0 + j * 16, 0, lane);
#pragma unroll
    for (int i = 0; i < 2; ++i) {
      v16h afrag = load_a_frag(A, K, wm0 + i * 16, k0, lane, M);
#pragma unroll
      for (int j = 0; j < 4; ++j)
        acc[i][j] = wmma16(afrag, bfrag[j], acc[i][j]);
    }
    __syncthreads();  // release buffer `cur` for overwrite at ks+2
  }

  // --- epilogue: v8f elem r -> row = tile_m + 8*(lane>>4) + r, col = tile_n + (lane&15)
#pragma unroll
  for (int i = 0; i < 2; ++i) {
#pragma unroll
    for (int j = 0; j < 4; ++j) {
      int col = n0 + wn0 + j * 16 + (lane & 15);
      int rb  = wm0 + i * 16 + ((lane >> 4) << 3);
      float bv = bias ? bias[col] : 0.f;
#pragma unroll
      for (int r = 0; r < 8; ++r) {
        int m = rb + r;
        if (m < M) {
          float val = acc[i][j][r] + bv;
          if (F16OUT)
            ((_Float16*)Cout)[(size_t)m * Nc + col] = (_Float16)val;
          else
            ((float*)Cout)[(size_t)m * Nc + col] = val;
        }
      }
    }
  }
}

// ---------------------------------------------------------------------------
// RoPE + split: qkv[B,N,3,H,64] (f16) -> q/k/v in [B,H,N,64] (f16).
// Token 0 (CLS) is passed through unrotated. One thread per (b,n,h,pair).
// ---------------------------------------------------------------------------
__global__ void rope_split_kernel(const _Float16* __restrict__ qkv,
                                  const float* __restrict__ rope,
                                  _Float16* __restrict__ qh,
                                  _Float16* __restrict__ kh,
                                  _Float16* __restrict__ vh,
                                  int B, int N, int H) {
  const int hd = 64;
  size_t idx = (size_t)blockIdx.x * blockDim.x + threadIdx.x;
  size_t total = (size_t)B * N * H * (hd / 2);
  if (idx >= total) return;
  int dp = (int)(idx & 31);
  size_t t = idx >> 5;
  int h = (int)(t % H); t /= H;
  int n = (int)(t % N);
  int b = (int)(t / N);
  int d0 = dp * 2, d1 = d0 + 1;

  const size_t C3 = (size_t)3 * H * hd;
  const _Float16* base = qkv + ((size_t)b * N + n) * C3 + (size_t)h * hd;
  float q0 = (float)base[d0],               q1 = (float)base[d1];
  float k0 = (float)base[H * hd + d0],      k1 = (float)base[H * hd + d1];
  float vv0 = (float)base[2 * H * hd + d0], vv1 = (float)base[2 * H * hd + d1];

  if (n >= 1) {  // NPT = 1
    const float* rp = rope + (size_t)(n - 1) * 2 * hd;  // [sin(64) | cos(64)]
    float s0 = rp[d0], s1 = rp[d1];
    float c0 = rp[hd + d0], c1 = rp[hd + d1];
    float nq0 = q0 * c0 - q1 * s0;   // rot(x): (x0,x1)->(-x1,x0)
    float nq1 = q1 * c1 + q0 * s1;
    float nk0 = k0 * c0 - k1 * s0;
    float nk1 = k1 * c1 + k0 * s1;
    q0 = nq0; q1 = nq1; k0 = nk0; k1 = nk1;
  }
  size_t o = (((size_t)b * H + h) * N + n) * hd;
  qh[o + d0] = (_Float16)q0;  qh[o + d1] = (_Float16)q1;
  kh[o + d0] = (_Float16)k0;  kh[o + d1] = (_Float16)k1;
  vh[o + d0] = (_Float16)vv0; vh[o + d1] = (_Float16)vv1;
}

// ---------------------------------------------------------------------------
// kv[bh][d][e] = sum_n K[bh][n][d] * V[bh][n][e]
// One block per (b,h); each of 8 waves reduces strided 32-token slabs into a
// full in-register 64x64 f32 accumulator (16 WMMA tiles), transposing K/V
// slabs through wave-private LDS. Final reduce: global f32 atomics.
// ---------------------------------------------------------------------------
__global__ __launch_bounds__(256) void kv_gemm_kernel(
    const _Float16* __restrict__ kh, const _Float16* __restrict__ vh,
    float* __restrict__ kv, int N) {
  const int bh = blockIdx.x;
  const int tid = threadIdx.x, wave = tid >> 5, lane = tid & 31;
  const _Float16* Kp = kh + (size_t)bh * N * 64;
  const _Float16* Vp = vh + (size_t)bh * N * 64;

  __shared__ __align__(128) _Float16 Kt[8][64 * 32];  // [d][n] per wave
  __shared__ __align__(128) _Float16 Vt[8][64 * 32];  // [e][n] per wave

  v8f acc[4][4];
  const v8f zac = {0.f, 0.f, 0.f, 0.f, 0.f, 0.f, 0.f, 0.f};
#pragma unroll
  for (int i = 0; i < 4; ++i)
#pragma unroll
    for (int j = 0; j < 4; ++j) acc[i][j] = zac;

  for (int nb = wave * 32; nb < N; nb += 256) {
    int n = nb + lane;
    if (n < N) {
      const v8h* kr = (const v8h*)(Kp + (size_t)n * 64);
      const v8h* vr = (const v8h*)(Vp + (size_t)n * 64);
#pragma unroll
      for (int c8 = 0; c8 < 8; ++c8) {
        v8h kk = kr[c8], vv = vr[c8];
#pragma unroll
        for (int e = 0; e < 8; ++e) {
          Kt[wave][(c8 * 8 + e) * 32 + lane] = kk[e];
          Vt[wave][(c8 * 8 + e) * 32 + lane] = vv[e];
        }
      }
    } else {
#pragma unroll
      for (int d = 0; d < 64; ++d) {
        Kt[wave][d * 32 + lane] = (_Float16)0.f;
        Vt[wave][d * 32 + lane] = (_Float16)0.f;
      }
    }
    asm volatile("" ::: "memory");  // wave-synchronous LDS: keep store->load order

    v16h bfrag[4];
#pragma unroll
    for (int j = 0; j < 4; ++j)
      bfrag[j] = load_bt_frag(Vt[wave], 32, j * 16, 0, lane);
#pragma unroll
    for (int i = 0; i < 4; ++i) {
      v16h afrag = load_a_frag(Kt[wave], 32, i * 16, 0, lane, 64);
#pragma unroll
      for (int j = 0; j < 4; ++j)
        acc[i][j] = wmma16(afrag, bfrag[j], acc[i][j]);
    }
    asm volatile("" ::: "memory");
  }

  float* dst = kv + (size_t)bh * 4096;
#pragma unroll
  for (int i = 0; i < 4; ++i)
#pragma unroll
    for (int j = 0; j < 4; ++j) {
      int col = j * 16 + (lane & 15);
      int rb  = i * 16 + ((lane >> 4) << 3);
#pragma unroll
      for (int r = 0; r < 8; ++r)
        atomicAdd(dst + (size_t)(rb + r) * 64 + col, acc[i][j][r]);
    }
}

// ---------------------------------------------------------------------------
// y[b,n,h*64+e] = scale * sum_d Q[bh][n][d] * kv[bh][d][e]   (f16 out)
// Block: 8 waves x 16 rows = 128 tokens; kv staged transposed in LDS.
// ---------------------------------------------------------------------------
__global__ __launch_bounds__(256) void q_kv_kernel(
    const _Float16* __restrict__ qh, const float* __restrict__ kv,
    _Float16* __restrict__ y, int N, int H, float scale) {
  const int bh = blockIdx.y;
  const int b = bh / H, h = bh % H;
  const int wave = threadIdx.x >> 5, lane = threadIdx.x & 31;
  const int m0 = blockIdx.x * 128 + wave * 16;
  const _Float16* Qp = qh + (size_t)bh * N * 64;

  __shared__ __align__(128) _Float16 kvt[64 * 64];  // kvt[e][d] = kv[d][e]
  {
    const float* src = kv + (size_t)bh * 4096;
    int base = threadIdx.x * 16;
#pragma unroll
    for (int i = 0; i < 16; ++i) {
      int idx = base + i;
      int e = idx >> 6, d = idx & 63;
      kvt[idx] = (_Float16)src[d * 64 + e];
    }
  }
  __syncthreads();

  v8f acc[4];
  const v8f zac = {0.f, 0.f, 0.f, 0.f, 0.f, 0.f, 0.f, 0.f};
#pragma unroll
  for (int j = 0; j < 4; ++j) acc[j] = zac;

#pragma unroll
  for (int k0 = 0; k0 < 64; k0 += 32) {
    v16h afrag = load_a_frag(Qp, 64, m0, k0, lane, N);
#pragma unroll
    for (int j = 0; j < 4; ++j) {
      v16h bfrag = load_bt_frag(kvt, 64, j * 16, k0, lane);
      acc[j] = wmma16(afrag, bfrag, acc[j]);
    }
  }

  const int C = H * 64;
#pragma unroll
  for (int j = 0; j < 4; ++j) {
    int col = h * 64 + j * 16 + (lane & 15);
    int rb  = m0 + ((lane >> 4) << 3);
#pragma unroll
    for (int r = 0; r < 8; ++r) {
      int n = rb + r;
      if (n < N)
        y[((size_t)b * N + n) * C + col] = (_Float16)(acc[j][r] * scale);
    }
  }
}

// ---------------------------------------------------------------------------
// Host launcher
// ---------------------------------------------------------------------------
extern "C" void kernel_launch(void* const* d_in, const int* in_sizes, int n_in,
                              void* d_out, int out_size, void* d_ws, size_t ws_size,
                              hipStream_t stream) {
  (void)in_sizes; (void)n_in; (void)out_size; (void)ws_size;

  const float* x      = (const float*)d_in[0];  // [8,4097,768]
  const float* rope   = (const float*)d_in[1];  // [4096,128]
  const float* qkv_w  = (const float*)d_in[2];  // [2304,768]
  const float* q_bias = (const float*)d_in[3];  // [768]
  const float* v_bias = (const float*)d_in[4];  // [768]
  const float* proj_w = (const float*)d_in[5];  // [768,768]
  const float* proj_b = (const float*)d_in[6];  // [768]
  float* out = (float*)d_out;

  const int B = 8, N = 4097, C = 768, H = 12, hd = 64;
  const int M = B * N;          // 32776
  const int C3 = 3 * C;         // 2304
  const int BH = B * H;         // 96

  // workspace carve (256B aligned regions)
  char* p = (char*)d_ws;
  auto carve = [&](size_t bytes) {
    void* r = (void*)p;
    p += (bytes + 255) & ~(size_t)255;
    return r;
  };
  _Float16* x_h    = (_Float16*)carve((size_t)M * C * 2);
  _Float16* qkvw_h = (_Float16*)carve((size_t)C3 * C * 2);
  _Float16* projw_h= (_Float16*)carve((size_t)C * C * 2);
  float*    bias3  = (float*)   carve((size_t)C3 * 4);
  _Float16* qkv_h  = (_Float16*)carve((size_t)M * C3 * 2);
  _Float16* q_h    = (_Float16*)carve((size_t)BH * N * hd * 2);
  _Float16* k_h    = (_Float16*)carve((size_t)BH * N * hd * 2);
  _Float16* v_h    = (_Float16*)carve((size_t)BH * N * hd * 2);
  float*    kvbuf  = (float*)   carve((size_t)BH * hd * hd * 4);
  _Float16* y_h    = (_Float16*)carve((size_t)M * C * 2);

  // 1) f32 -> f16 conversions
  cvt_f32_f16_kernel<<<4096, 256, 0, stream>>>(x, x_h, (size_t)M * C);
  cvt_f32_f16_kernel<<<2048, 256, 0, stream>>>(qkv_w, qkvw_h, (size_t)C3 * C);
  cvt_f32_f16_kernel<<<1024, 256, 0, stream>>>(proj_w, projw_h, (size_t)C * C);
  build_bias_kernel<<<(3 * C + 255) / 256, 256, 0, stream>>>(q_bias, v_bias, bias3, C);

  // 2) QKV GEMM (f16 out, fused bias): [M,2304] = x_h @ qkvw^T + bias3
  {
    dim3 grid((M + 127) / 128, C3 / 128);
    gemm_xWT_kernel<true><<<grid, 256, 0, stream>>>(x_h, qkvw_h, bias3, qkv_h, M, C3, C);
  }

  // 3) RoPE + split into [B,H,N,64] f16
  {
    size_t total = (size_t)B * N * H * (hd / 2);
    rope_split_kernel<<<(unsigned)((total + 255) / 256), 256, 0, stream>>>(
        qkv_h, rope, q_h, k_h, v_h, B, N, H);
  }

  // 4) kv = K^T V per head (zero, then WMMA + atomics)
  zero_f32_kernel<<<1536, 256, 0, stream>>>(kvbuf, (size_t)BH * hd * hd);
  kv_gemm_kernel<<<BH, 256, 0, stream>>>(k_h, v_h, kvbuf, N);

  // 5) y = (Q @ kv) / (hd * N), back to [B,N,C] f16
  {
    dim3 grid((N + 127) / 128, BH);
    float scale = 1.0f / ((float)hd * (float)N);
    q_kv_kernel<<<grid, 256, 0, stream>>>(q_h, kvbuf, y_h, N, H, scale);
  }

  // 6) final projection (f32 out to d_out): out = y @ proj_w^T + proj_b
  {
    dim3 grid((M + 127) / 128, C / 128);
    gemm_xWT_kernel<false><<<grid, 256, 0, stream>>>(y_h, projw_h, proj_b, out, M, C, C);
  }
}
